// VAE_1649267441893
// MI455X (gfx1250) — matrix operand order repaired and checked by reading, not compiled, not run on hardware
//
#include <hip/hip_runtime.h>
#include <math.h>

// ---------------------------------------------------------------------------
// CW-VAE loss on MI455X (gfx1250, wave32).
//   - 3x (8192 x 8192 x 784) + 1x (8192 x 8192 x 64) Gram matrices via
//     v_wmma_f32_16x16x32_bf16, fused with rsqrt(c0 + c1*d) reduction.
//   - Double-buffered LDS tiles fed by global_load_async_to_lds_b128
//     (ASYNCcnt pipeline), VGPR-staged fallback if builtin unavailable.
//   - Per-block partial sums -> double atomics -> 1-thread finalize.
// Workspace: ~27.4 MB.
// ---------------------------------------------------------------------------

typedef __attribute__((ext_vector_type(16))) __bf16 v16bf;
typedef __attribute__((ext_vector_type(8)))  float  v8f;
typedef int v4i __attribute__((vector_size(16)));   // matches builtin param type

#define NROW   8192
#define TILE   128
#define LDSLD  40          // bf16 per LDS row: 32 data + 8 pad (conflict-free)
#define PI_D   3.14159265358979323846

#if defined(__AMDGCN__) && __has_builtin(__builtin_amdgcn_global_load_async_to_lds_b128) && \
    __has_builtin(__builtin_amdgcn_s_wait_asynccnt)
#define USE_ASYNC_LDS 1
#else
#define USE_ASYNC_LDS 0
#endif

#define AS1 __attribute__((address_space(1)))
#define AS3 __attribute__((address_space(3)))

// Generic->global: same 64-bit VA. Generic->LDS: low 32 bits of a generic LDS
// pointer are the LDS byte offset (flat aperture rules), so build the AS3
// pointer from them via integer casts (avoids addrspacecast legality issues).
static __device__ inline AS1 v4i* as_global_v(const void* p) {
    return (AS1 v4i*)(unsigned long long)p;
}
static __device__ inline AS3 v4i* as_lds_v(const void* p) {
    return (AS3 v4i*)(unsigned int)(unsigned long long)p;
}

union Frag16 { v16bf v; uint4 q[2]; };

// ---- convert fp32 -> bf16 with K padding (pad region written as 0) --------
__global__ void cvt_pad_kernel(const float* __restrict__ src,
                               __bf16* __restrict__ dst,
                               int rows, int kin, int kout) {
    int total = rows * kout;
    for (int i = blockIdx.x * blockDim.x + threadIdx.x; i < total;
         i += gridDim.x * blockDim.x) {
        int r = i / kout;
        int k = i - r * kout;
        float v = (k < kin) ? src[r * kin + k] : 0.0f;
        dst[i] = (__bf16)v;
    }
}

// ---- per-row squared norms from the bf16 copies (one wave per row) --------
__global__ void row_norms_kernel(const __bf16* __restrict__ src, int ld,
                                 float* __restrict__ n2,
                                 double* __restrict__ szz) {
    int lane = threadIdx.x & 31;
    int wave = threadIdx.x >> 5;
    int row  = blockIdx.x * 8 + wave;
    float s = 0.0f;
    for (int k = lane; k < ld; k += 32) {
        float v = (float)src[(size_t)row * ld + k];
        s = fmaf(v, v, s);
    }
    #pragma unroll
    for (int off = 16; off > 0; off >>= 1) s += __shfl_down(s, off, 32);
    if (lane == 0) {
        n2[row] = s;
        if (szz) atomicAdd(szz, (double)s);   // total sum(z*z) for B-term
    }
}

__global__ void zero_acc_kernel(double* S) {
    if (threadIdx.x < 5) S[threadIdx.x] = 0.0;
}

// ---- main fused Gram + phi-reduction kernel -------------------------------
// grid = (64, 64, 4): z selects combo {data*data, rec*rec, data*rec, z*z}
__global__ __launch_bounds__(256)
void gram_phi_kernel(const __bf16* __restrict__ dbf,
                     const __bf16* __restrict__ rbf,
                     const __bf16* __restrict__ zbf,
                     const float*  __restrict__ dn2,
                     const float*  __restrict__ rn2,
                     const float*  __restrict__ zn2,
                     double* __restrict__ S,
                     float c1s, float c1n, float yf) {
    __shared__ __align__(16) __bf16 ldsA[2][TILE * LDSLD];
    __shared__ __align__(16) __bf16 ldsB[2][TILE * LDSLD];
    __shared__ float ldsXn[TILE];
    __shared__ float ldsYn[TILE];
    __shared__ float red[256];

    const int combo = blockIdx.z;
    const __bf16 *X, *Y; const float *xn2, *yn2;
    int K, ld; float c0, c1;
    if (combo == 0)      { X=dbf; Y=dbf; xn2=dn2; yn2=dn2; K=800; ld=800; c0=1.0f; c1=c1s; }
    else if (combo == 1) { X=rbf; Y=rbf; xn2=rn2; yn2=rn2; K=800; ld=800; c0=1.0f; c1=c1s; }
    else if (combo == 2) { X=dbf; Y=rbf; xn2=dn2; yn2=rn2; K=800; ld=800; c0=1.0f; c1=c1s; }
    else                 { X=zbf; Y=zbf; xn2=zn2; yn2=zn2; K=64;  ld=64;  c0=yf;   c1=c1n; }

    const int tid  = threadIdx.x;
    const int lane = tid & 31;
    const int wave = tid >> 5;
    const int wm   = wave >> 1;       // 0..3 : 32-row slab of M
    const int wn   = wave & 1;        // 0..1 : 64-col slab of N
    const int m16  = lane & 15;
    const int hi   = lane >> 4;

    const int i0 = blockIdx.y * TILE;
    const int j0 = blockIdx.x * TILE;

    if (tid < TILE) ldsXn[tid]        = xn2[i0 + tid];
    else            ldsYn[tid - TILE] = yn2[j0 + (tid - TILE)];

    v8f c[2][4] = {};

    const int lrow = tid >> 2;         // 0..63
    const int lk   = (tid & 3) * 8;    // 0,8,16,24 (bf16 units, 16B chunks)
    const int nk   = K >> 5;           // K / 32 steps

#if USE_ASYNC_LDS
    // issue one 128x32 A tile + B tile into LDS buffer `buf` (4 b128/thread)
    auto issue_tile = [&](int buf, int kb) {
        #pragma unroll
        for (int it = 0; it < 2; ++it) {
            int row = lrow + it * 64;
            __builtin_amdgcn_global_load_async_to_lds_b128(
                as_global_v(X + (size_t)(i0 + row) * ld + kb + lk),
                as_lds_v(&ldsA[buf][row * LDSLD + lk]), 0, 0);
            __builtin_amdgcn_global_load_async_to_lds_b128(
                as_global_v(Y + (size_t)(j0 + row) * ld + kb + lk),
                as_lds_v(&ldsB[buf][row * LDSLD + lk]), 0, 0);
        }
    };
    issue_tile(0, 0);
#endif

    for (int kbi = 0; kbi < nk; ++kbi) {
        const int cur = kbi & 1;
        const int kb  = kbi << 5;
#if USE_ASYNC_LDS
        __builtin_amdgcn_s_wait_asynccnt(0);  // own async writes to `cur` done
        __syncthreads();                      // everyone's done; buf cur^1 free
        if (kbi + 1 < nk) issue_tile(cur ^ 1, kb + 32);  // overlaps with WMMA
#else
        __syncthreads();
        {   // batch loads into temps, then store (avoids serialized waits)
            uint4 ta[2], tb[2];
            #pragma unroll
            for (int it = 0; it < 2; ++it) {
                int row = lrow + it * 64;
                ta[it] = *(const uint4*)(X + (size_t)(i0 + row) * ld + kb + lk);
                tb[it] = *(const uint4*)(Y + (size_t)(j0 + row) * ld + kb + lk);
            }
            #pragma unroll
            for (int it = 0; it < 2; ++it) {
                int row = lrow + it * 64;
                *(uint4*)(&ldsA[cur][row * LDSLD + lk]) = ta[it];
                *(uint4*)(&ldsB[cur][row * LDSLD + lk]) = tb[it];
            }
        }
        if (kb + 32 < K) {  // pull next K-chunk toward L2 (global_prefetch_b8)
            __builtin_prefetch(X + (size_t)(i0 + lrow) * ld + kb + 32, 0, 3);
            __builtin_prefetch(Y + (size_t)(j0 + lrow) * ld + kb + 32, 0, 3);
        }
        __syncthreads();
#endif
        // A fragment (16-bit A 16x32): lane m=L%16, hi=L/16 holds
        //   K = {hi*8..hi*8+7, 16+hi*8..16+hi*8+7}
        Frag16 a[2];
        #pragma unroll
        for (int rm = 0; rm < 2; ++rm) {
            const __bf16* p = &ldsA[cur][(wm * 32 + rm * 16 + m16) * LDSLD + hi * 8];
            a[rm].q[0] = *(const uint4*)p;
            a[rm].q[1] = *(const uint4*)(p + 16);
        }
        // B fragment (16-bit B 32x16): lane n=L%16 holds K = {hi*8..hi*8+7, 16+hi*8..16+hi*8+7}
        Frag16 b[4];
        #pragma unroll
        for (int rn = 0; rn < 4; ++rn) {
            const __bf16* p = &ldsB[cur][(wn * 64 + rn * 16 + m16) * LDSLD + hi * 8];
            b[rn].q[0] = *(const uint4*)p;
            b[rn].q[1] = *(const uint4*)(p + 16);
        }
        #pragma unroll
        for (int rm = 0; rm < 2; ++rm)
            #pragma unroll
            for (int rn = 0; rn < 4; ++rn)
                c[rm][rn] = __builtin_amdgcn_wmma_f32_16x16x32_bf16(
                    false, a[rm].v, false, b[rn].v,
                    (short)0, c[rm][rn], false, false);
    }

    // Epilogue: C/D layout -> vgpr v holds M = v + 8*hi, N = lane%16.
    // d = max(|x|^2 + |y|^2 - 2g, 0); accumulate rsqrt(c0 + c1*d).
    float lsum = 0.0f;
    #pragma unroll
    for (int rm = 0; rm < 2; ++rm) {
        #pragma unroll
        for (int rn = 0; rn < 4; ++rn) {
            float yn = ldsYn[wn * 64 + rn * 16 + m16];
            #pragma unroll
            for (int v = 0; v < 8; ++v) {
                float xn = ldsXn[wm * 32 + rm * 16 + hi * 8 + v];
                float d  = fmaxf(xn + yn - 2.0f * c[rm][rn][v], 0.0f);
                lsum += rsqrtf(fmaf(c1, d, c0));
            }
        }
    }

    red[tid] = lsum;
    __syncthreads();
    #pragma unroll
    for (int s = 128; s > 0; s >>= 1) {
        if (tid < s) red[tid] += red[tid + s];
        __syncthreads();
    }
    if (tid == 0) atomicAdd(&S[combo], (double)red[0]);
}

// ---- combine the four reductions into the scalar loss ---------------------
__global__ void finalize_kernel(const double* __restrict__ S,
                                float* __restrict__ out, double y) {
    double cnt = (double)NROW * (double)NROW;
    double A = S[0] / cnt, B = S[1] / cnt, C = S[2] / cnt;
    double T = 1.0 / (2.0 * sqrt(PI_D * y));
    double l_rec = log(T * (A + B - 2.0 * C));
    double k1z = 1.0 / 125.0;                      // 1/(2*64-3)
    double Az  = S[3] / cnt;
    double Bz  = 2.0 / sqrt(y + 0.5 + k1z * S[4]); // S[4] = sum(z*z)
    double l_z = log(1.0 / sqrt(1.0 + y) + Az - Bz);
    out[0] = (float)(l_rec + l_z);
}

extern "C" void kernel_launch(void* const* d_in, const int* in_sizes, int n_in,
                              void* d_out, int out_size, void* d_ws, size_t ws_size,
                              hipStream_t stream) {
    const float* data = (const float*)d_in[0];   // 8192 x 784
    const float* rec  = (const float*)d_in[1];   // 8192 x 784
    const float* z    = (const float*)d_in[2];   // 8192 x 64

    // workspace layout (all 16B aligned)
    const size_t off_d  = 0;                       // 8192*800 bf16 = 13,107,200
    const size_t off_r  = 13107200;
    const size_t off_z  = 26214400;                // 8192*64 bf16 = 1,048,576
    const size_t off_dn = 27262976;                // 8192 f32
    const size_t off_rn = 27295744;
    const size_t off_zn = 27328512;
    const size_t off_S  = 27361280;                // 5 doubles
    if (ws_size < off_S + 5 * sizeof(double)) return;  // insufficient scratch

    char* ws = (char*)d_ws;
    __bf16* dbf = (__bf16*)(ws + off_d);
    __bf16* rbf = (__bf16*)(ws + off_r);
    __bf16* zbf = (__bf16*)(ws + off_z);
    float*  dn2 = (float*)(ws + off_dn);
    float*  rn2 = (float*)(ws + off_rn);
    float*  zn2 = (float*)(ws + off_zn);
    double* S   = (double*)(ws + off_S);

    const double y = pow(4.0 / (3.0 * 8192.0), 0.4);   // Silverman bandwidth
    const float c1s = (float)(1.0 / (1565.0 * y));     // sampling: rsqrt(1 + d/(1565 y))
    const float c1n = 1.0f / 125.0f;                   // normality: rsqrt(y + d/125)
    const float yf  = (float)y;

    zero_acc_kernel<<<1, 32, 0, stream>>>(S);
    cvt_pad_kernel<<<4096, 256, 0, stream>>>(data, dbf, 8192, 784, 800);
    cvt_pad_kernel<<<4096, 256, 0, stream>>>(rec,  rbf, 8192, 784, 800);
    cvt_pad_kernel<<<2048, 256, 0, stream>>>(z,    zbf, 8192, 64,  64);
    row_norms_kernel<<<1024, 256, 0, stream>>>(dbf, 800, dn2, nullptr);
    row_norms_kernel<<<1024, 256, 0, stream>>>(rbf, 800, rn2, nullptr);
    row_norms_kernel<<<1024, 256, 0, stream>>>(zbf, 64,  zn2, S + 4);
    gram_phi_kernel<<<dim3(64, 64, 4), 256, 0, stream>>>(
        dbf, rbf, zbf, dn2, rn2, zn2, S, c1s, c1n, yf);
    finalize_kernel<<<1, 1, 0, stream>>>(S, (float*)d_out, y);
}
